// GNBlock_68633577390246
// MI455X (gfx1250) — matrix operand order repaired and checked by reading, not compiled, not run on hardware
//
#include <hip/hip_runtime.h>
#include <cmath>

#define HOPS  3
#define NHEAD 4
#define F_IN  128
#define F_OUT 128
#define E_IN  64
#define HF    512      // NHEAD * F_OUT
#define SLOPE 0.2f

typedef __attribute__((ext_vector_type(16))) __bf16 v16bf;
typedef __attribute__((ext_vector_type(8)))  float  v8f;

union BFrag { v16bf v; uint4 q[2]; };

static __device__ __forceinline__ unsigned short f2bf(float f) {
  unsigned int u = __float_as_uint(f);
  u += 0x7FFFu + ((u >> 16) & 1u);            // round-to-nearest-even
  return (unsigned short)(u >> 16);
}

// ---------------- conversion / fill helpers ----------------
__global__ void k_f2bf(const float* __restrict__ in, unsigned short* __restrict__ out, int n) {
  int i = blockIdx.x * blockDim.x + threadIdx.x;
  if (i < n) out[i] = f2bf(in[i]);
}

// in: [K, Ncols] f32 row-major  ->  out: [Ncols, K] bf16 row-major (i.e. B^T)
__global__ void k_transpose_f2bf(const float* __restrict__ in, unsigned short* __restrict__ out,
                                 int K, int Ncols) {
  int i = blockIdx.x * blockDim.x + threadIdx.x;
  if (i >= K * Ncols) return;
  int n = i / K, k = i - n * K;
  out[i] = f2bf(in[(size_t)k * Ncols + n]);
}

__global__ void k_fill(float* __restrict__ p, float v, int n) {
  int i = blockIdx.x * blockDim.x + threadIdx.x;
  if (i < n) p[i] = v;
}

// Ballt[j][k] bf16, [16][64]; j = hop*NHEAD + h (rows >= 12 zero-padded)
// Ballt[j][k] = sum_f Wes[hop][k, h*F_OUT+f] * a_edge[hop][h][f]
__global__ void k_ball(const float* __restrict__ Wes, const float* __restrict__ a_edge,
                       unsigned short* __restrict__ Ballt) {
  int tid = threadIdx.x;                       // 1024 threads: j in [0,16), k in [0,64)
  int j = tid >> 6, k = tid & 63;
  float s = 0.f;
  if (j < HOPS * NHEAD) {
    int hop = j >> 2, h = j & 3;
    const float* wrow = Wes + ((size_t)hop * E_IN + k) * HF + h * F_OUT;
    const float* av   = a_edge + (size_t)hop * NHEAD * F_OUT + h * F_OUT;
    for (int f = 0; f < F_OUT; ++f) s += wrow[f] * av[f];
  }
  Ballt[j * E_IN + k] = f2bf(s);
}

// ---------------- WMMA bf16 GEMM ----------------
// C[M,Nc] f32 = A[M,K] (bf16 row-major) * B[K,Nc] given as Bt[Nc,K] (bf16 row-major)
// Requires M%16==0, Nc%16==0, K%32==0. One wave computes a 16x16 tile.
__global__ void k_gemm_bf16(const unsigned short* __restrict__ A,
                            const unsigned short* __restrict__ Bt,
                            float* __restrict__ C, int M, int Nc, int K) {
  int wave = (int)((blockIdx.x * blockDim.x + threadIdx.x) >> 5);
  int lane = threadIdx.x & 31;
  int tilesN = Nc >> 4;
  int tiles  = (M >> 4) * tilesN;
  if (wave >= tiles) return;
  int tm = wave / tilesN, tn = wave - tm * tilesN;
  int half = lane >> 4, m = lane & 15;

  v8f acc = {};
  // A frag: elems 0..7 -> K = k0 + half*8 + e ; elems 8..15 -> K = k0 + 16 + half*8 + e
  const unsigned short* arow = A  + (size_t)(tm * 16 + m) * K + half * 8;
  // B frag: elems 0..7 -> K = k0 + half*8 + e ; elems 8..15 -> K = k0 + 16 + half*8 + e (Bt[col][K])
  const unsigned short* brow = Bt + (size_t)(tn * 16 + m) * K + half * 8;
  for (int k0 = 0; k0 < K; k0 += 32) {
    BFrag a, b;
    a.q[0] = *(const uint4*)(arow + k0);
    a.q[1] = *(const uint4*)(arow + k0 + 16);
    b.q[0] = *(const uint4*)(brow + k0);
    b.q[1] = *(const uint4*)(brow + k0 + 16);
    acc = __builtin_amdgcn_wmma_f32_16x16x32_bf16(false, a.v, false, b.v,
                                                  (short)0, acc, false, false);
  }
  // C/D layout: VGPR r -> M = half*8 + r, N = m
  float* crow = C + (size_t)(tm * 16 + half * 8) * Nc + tn * 16 + m;
#pragma unroll
  for (int r = 0; r < 8; ++r) crow[(size_t)r * Nc] = acc[r];
}

// ---------------- attention pipeline ----------------
// s_src[n,h] = h[n,h,:]·a_src[h,:] ; s_dst likewise (a_* already offset per hop)
__global__ void k_ssd(const float* __restrict__ hbuf, const float* __restrict__ asv,
                      const float* __restrict__ adv, float* __restrict__ ssrc,
                      float* __restrict__ sdst, int N) {
  int t = blockIdx.x * blockDim.x + threadIdx.x;
  if (t >= N * NHEAD) return;
  int n = t >> 2, h = t & 3;
  const float* hr = hbuf + (size_t)n * HF + h * F_OUT;
  const float* as = asv + h * F_OUT;
  const float* ad = adv + h * F_OUT;
  float ss = 0.f, sd = 0.f;
  for (int f = 0; f < F_OUT; ++f) { float v = hr[f]; ss += v * as[f]; sd += v * ad[f]; }
  ssrc[t] = ss; sdst[t] = sd;
}

static __device__ __forceinline__ void atomicMaxF(float* addr, float v) {
  if (v >= 0.f) atomicMax((int*)addr, __float_as_int(v));
  else          atomicMin((unsigned int*)addr, __float_as_uint(v));
}

__global__ void k_logits(const int* __restrict__ src, const int* __restrict__ dst,
                         const float* __restrict__ ssrc, const float* __restrict__ sdst,
                         const float* __restrict__ alpha_all, int hop,
                         float* __restrict__ w_edge, float* __restrict__ amax, int E) {
  int e = blockIdx.x * blockDim.x + threadIdx.x;
  if (e >= E) return;
  int s = src[e], d = dst[e];
#pragma unroll
  for (int h = 0; h < NHEAD; ++h) {
    float a = ssrc[s * NHEAD + h] + sdst[d * NHEAD + h] +
              alpha_all[(size_t)e * 16 + hop * NHEAD + h];
    a = (a > 0.f) ? a : SLOPE * a;
    w_edge[(size_t)e * NHEAD + h] = a;
    atomicMaxF(&amax[d * NHEAD + h], a);
  }
}

__global__ void k_expw(const int* __restrict__ dst, const float* __restrict__ amax,
                       float* __restrict__ w_edge, float* __restrict__ denom, int E) {
  int e = blockIdx.x * blockDim.x + threadIdx.x;
  if (e >= E) return;
  int d = dst[e];
#pragma unroll
  for (int h = 0; h < NHEAD; ++h) {
    float w = __expf(w_edge[(size_t)e * NHEAD + h] - amax[d * NHEAD + h]);
    w_edge[(size_t)e * NHEAD + h] = w;
    atomicAdd(&denom[d * NHEAD + h], w);
  }
}

// one block (128 threads) per edge: acc[dst,h,f] += w[e,h] * h[src,h,f]
__global__ void k_scatter(const int* __restrict__ src, const int* __restrict__ dst,
                          const float* __restrict__ w_edge, const float* __restrict__ hbuf,
                          float* __restrict__ acc) {
  int e = blockIdx.x;
  int f = threadIdx.x;
  int s = src[e], d = dst[e];
  const float* hrow = hbuf + (size_t)s * HF;
  float* arow = acc + (size_t)d * HF;
#pragma unroll
  for (int h = 0; h < NHEAD; ++h)
    atomicAdd(&arow[h * F_OUT + f], w_edge[(size_t)e * NHEAD + h] * hrow[h * F_OUT + f]);
}

__global__ void k_finalize(const float* __restrict__ acc, const float* __restrict__ denom,
                           float* __restrict__ C, int N) {
  int t = blockIdx.x * blockDim.x + threadIdx.x;
  if (t >= N * F_OUT) return;
  int n = t >> 7, f = t & 127;
  float s = 0.f;
#pragma unroll
  for (int h = 0; h < NHEAD; ++h)
    s += acc[(size_t)n * HF + h * F_OUT + f] / denom[n * NHEAD + h];
  C[t] = s * (1.f / NHEAD);
}

__global__ void k_setx(const float* __restrict__ Cin, float* __restrict__ X,
                       unsigned short* __restrict__ xb, int n) {
  int i = blockIdx.x * blockDim.x + threadIdx.x;
  if (i < n) { float v = Cin[i]; X[i] = v; xb[i] = f2bf(v); }
}

__global__ void k_gate(const float* __restrict__ G1, const float* __restrict__ G2,
                       const float* __restrict__ Cin, float* __restrict__ X,
                       unsigned short* __restrict__ xb, int n) {
  int i = blockIdx.x * blockDim.x + threadIdx.x;
  if (i >= n) return;
  float g = 1.f / (1.f + __expf(-(G1[i] + G2[i])));
  float v = g * Cin[i] + (1.f - g) * X[i];
  X[i] = v; xb[i] = f2bf(v);
}

// ---------------- driver ----------------
extern "C" void kernel_launch(void* const* d_in, const int* in_sizes, int n_in,
                              void* d_out, int out_size, void* d_ws, size_t ws_size,
                              hipStream_t stream) {
  const float* node_fts = (const float*)d_in[0];
  const float* edge_fts = (const float*)d_in[1];
  const int*   edges    = (const int*)  d_in[2];
  const float* Ws       = (const float*)d_in[3];
  const float* Wes      = (const float*)d_in[4];
  const float* a_src    = (const float*)d_in[5];
  const float* a_dst    = (const float*)d_in[6];
  const float* a_edge   = (const float*)d_in[7];
  const float* W_prior  = (const float*)d_in[8];
  const float* W_curr   = (const float*)d_in[9];

  const int N = in_sizes[0] / F_IN;
  const int E = in_sizes[2] / 2;
  const int* src = edges;
  const int* dst = edges + E;

  char* p = (char*)d_ws;
  auto alloc = [&](size_t bytes) -> char* {
    char* r = p; p += (bytes + 255) & ~(size_t)255; return r;
  };
  unsigned short* xb    = (unsigned short*)alloc((size_t)N * F_IN * 2);
  unsigned short* cb    = (unsigned short*)alloc((size_t)N * F_OUT * 2);
  unsigned short* Wb    = (unsigned short*)alloc((size_t)HOPS * HF * F_IN * 2);
  unsigned short* Wpb   = (unsigned short*)alloc((size_t)F_OUT * F_OUT * 2);
  unsigned short* Wcb   = (unsigned short*)alloc((size_t)F_OUT * F_OUT * 2);
  unsigned short* Ballt = (unsigned short*)alloc(16 * E_IN * 2);
  unsigned short* efb   = (unsigned short*)alloc((size_t)E * E_IN * 2);
  float* alpha_all = (float*)alloc((size_t)E * 16 * 4);
  float* hbuf      = (float*)alloc((size_t)N * HF * 4);
  float* ssrc      = (float*)alloc((size_t)N * NHEAD * 4);
  float* sdst      = (float*)alloc((size_t)N * NHEAD * 4);
  float* amax      = (float*)alloc((size_t)N * NHEAD * 4);
  float* denom     = (float*)alloc((size_t)N * NHEAD * 4);
  float* w_edge    = (float*)alloc((size_t)E * NHEAD * 4);
  float* accb      = (float*)alloc((size_t)N * HF * 4);
  float* Cf        = (float*)alloc((size_t)N * F_OUT * 4);
  float* Xf        = (float*)alloc((size_t)N * F_OUT * 4);
  float* G1        = (float*)alloc((size_t)N * F_OUT * 4);
  float* G2        = (float*)alloc((size_t)N * F_OUT * 4);

  auto cdiv = [](int a, int b) { return (a + b - 1) / b; };

  // one-time conversions (bf16 + weight transposes)
  k_f2bf<<<cdiv(N * F_IN, 256), 256, 0, stream>>>(node_fts, xb, N * F_IN);
  for (int i = 0; i < HOPS; ++i)
    k_transpose_f2bf<<<cdiv(HF * F_IN, 256), 256, 0, stream>>>(
        Ws + (size_t)i * F_IN * HF, Wb + (size_t)i * HF * F_IN, F_IN, HF);
  k_transpose_f2bf<<<cdiv(F_OUT * F_OUT, 256), 256, 0, stream>>>(W_prior, Wpb, F_OUT, F_OUT);
  k_transpose_f2bf<<<cdiv(F_OUT * F_OUT, 256), 256, 0, stream>>>(W_curr,  Wcb, F_OUT, F_OUT);
  k_f2bf<<<cdiv(E * E_IN, 256), 256, 0, stream>>>(edge_fts, efb, E * E_IN);
  k_ball<<<1, 1024, 0, stream>>>(Wes, a_edge, Ballt);

  // all-hop edge attention terms: [E,16] = edge_fts @ Ball  (WMMA)
  { int tiles = (E / 16) * 1;
    k_gemm_bf16<<<cdiv(tiles * 32, 256), 256, 0, stream>>>(efb, Ballt, alpha_all, E, 16, E_IN); }

  for (int hop = 0; hop < HOPS; ++hop) {
    // h = x @ W[hop]  (WMMA, [N,512])
    { int tiles = (N / 16) * (HF / 16);
      k_gemm_bf16<<<cdiv(tiles * 32, 256), 256, 0, stream>>>(
          xb, Wb + (size_t)hop * HF * F_IN, hbuf, N, HF, F_IN); }
    k_ssd<<<cdiv(N * NHEAD, 256), 256, 0, stream>>>(
        hbuf, a_src + (size_t)hop * NHEAD * F_OUT, a_dst + (size_t)hop * NHEAD * F_OUT,
        ssrc, sdst, N);
    k_fill<<<cdiv(N * NHEAD, 256), 256, 0, stream>>>(amax, -INFINITY, N * NHEAD);
    k_fill<<<cdiv(N * NHEAD, 256), 256, 0, stream>>>(denom, 1e-16f, N * NHEAD);
    k_fill<<<cdiv(N * HF, 256), 256, 0, stream>>>(accb, 0.f, N * HF);

    k_logits<<<cdiv(E, 256), 256, 0, stream>>>(src, dst, ssrc, sdst, alpha_all, hop,
                                               w_edge, amax, E);
    k_expw<<<cdiv(E, 256), 256, 0, stream>>>(dst, amax, w_edge, denom, E);
    k_scatter<<<E, 128, 0, stream>>>(src, dst, w_edge, hbuf, accb);
    k_finalize<<<cdiv(N * F_OUT, 256), 256, 0, stream>>>(accb, denom, Cf, N);

    if (hop == 0) {
      k_setx<<<cdiv(N * F_OUT, 256), 256, 0, stream>>>(Cf, Xf, xb, N * F_OUT);
    } else {
      k_f2bf<<<cdiv(N * F_OUT, 256), 256, 0, stream>>>(Cf, cb, N * F_OUT);
      int tiles = (N / 16) * (F_OUT / 16);
      // G1 = prior @ W_prior (xb still holds bf16 of previous x), G2 = curr @ W_curr
      k_gemm_bf16<<<cdiv(tiles * 32, 256), 256, 0, stream>>>(xb, Wpb, G1, N, F_OUT, F_OUT);
      k_gemm_bf16<<<cdiv(tiles * 32, 256), 256, 0, stream>>>(cb, Wcb, G2, N, F_OUT, F_OUT);
      k_gate<<<cdiv(N * F_OUT, 256), 256, 0, stream>>>(G1, G2, Cf, Xf, xb, N * F_OUT);
    }
  }

  hipMemcpyAsync(d_out, Xf, (size_t)N * F_OUT * sizeof(float),
                 hipMemcpyDeviceToDevice, stream);
}
